// CrossModalAlignment_32126355374759
// MI455X (gfx1250) — matrix operand order repaired and checked by reading, not compiled, not run on hardware
//
#include <hip/hip_runtime.h>

// ---------------------------------------------------------------------------
// CDNA5 / gfx1250 implementation: all GEMMs via v_wmma_f32_16x16x32_bf16.
// wave32, 256-thread blocks = 8 waves, 64x128 block tile, 32x32 per wave.
// ---------------------------------------------------------------------------

typedef __attribute__((ext_vector_type(16))) __bf16 v16bf;
typedef __attribute__((ext_vector_type(8)))  __bf16 v8bf;
typedef __attribute__((ext_vector_type(8)))  float  v8f;

#define LDK 40              // padded LDS row (32 data + 8 pad) to avoid bank conflicts
#define NEGV (-1.0e9f)

static constexpr int Bc = 16;
static constexpr int Mc = 1024;
static constexpr int Nc = 1024;
static constexpr int Dc = 768;

// ---------------- WMMA fragment loads (wave32 ISA layouts) ------------------
// A (16x32 bf16): lane<16 holds row=lane, K 0..7 & 16..23; lane>=16 same row,
// K 8..15 & 24..31.  LDS tile stored row-major [m][k], ld = LDK.
__device__ __forceinline__ v16bf frag_a(const __bf16* As, int lane) {
    int r  = lane & 15;
    int kb = (lane < 16) ? 0 : 8;
    const __bf16* p = As + r * LDK + kb;
    v8bf lo = *(const v8bf*)(p);
    v8bf hi = *(const v8bf*)(p + 16);
    return __builtin_shufflevector(lo, hi, 0,1,2,3,4,5,6,7,8,9,10,11,12,13,14,15);
}

// B (32x16 bf16): lane<16 holds col=lane, K 0..7 & 16..23; lane>=16 col=lane-16, K 8..15 & 24..31.
// LDS tile stored n-major [n][k], ld = LDK.
__device__ __forceinline__ v16bf frag_b(const __bf16* Bs, int lane) {
    int c  = lane & 15;
    int kb = (lane < 16) ? 0 : 8;
    const __bf16* p = Bs + c * LDK + kb;
    v8bf lo = *(const v8bf*)(p);
    v8bf hi = *(const v8bf*)(p + 16);
    return __builtin_shufflevector(lo, hi, 0,1,2,3,4,5,6,7,8,9,10,11,12,13,14,15);
}

__device__ __forceinline__ void mma_step(const __bf16* As, const __bf16* Bs,
                                         int wm, int wn, int lane, v8f acc[4]) {
    v16bf a0 = frag_a(As + (wm * 32     ) * LDK, lane);
    v16bf a1 = frag_a(As + (wm * 32 + 16) * LDK, lane);
    v16bf b0 = frag_b(Bs + (wn * 32     ) * LDK, lane);
    v16bf b1 = frag_b(Bs + (wn * 32 + 16) * LDK, lane);
    acc[0] = __builtin_amdgcn_wmma_f32_16x16x32_bf16(false, a0, false, b0, (short)0, acc[0], false, false);
    acc[1] = __builtin_amdgcn_wmma_f32_16x16x32_bf16(false, a0, false, b1, (short)0, acc[1], false, false);
    acc[2] = __builtin_amdgcn_wmma_f32_16x16x32_bf16(false, a1, false, b0, (short)0, acc[2], false, false);
    acc[3] = __builtin_amdgcn_wmma_f32_16x16x32_bf16(false, a1, false, b1, (short)0, acc[3], false, false);
}

// ---------------- LDS staging helpers --------------------------------------
// A tile: 64 rows x 32 k from bf16 global [m][k]; 256 thr * 8 elems = 1 b128 each.
__device__ __forceinline__ void stage_a_bf16(__bf16* As, const __bf16* gA, int lda, int tid) {
    int e = tid * 8;
    int r = e >> 5, k = e & 31;
    v8bf v = *(const v8bf*)(gA + (size_t)r * lda + k);
    *(v8bf*)(As + r * LDK + k) = v;
}

// A tile from fp32 global (convert to bf16 while staging).
__device__ __forceinline__ void stage_a_f32(__bf16* As, const float* gA, int lda, int tid) {
    int e = tid * 8;
    int r = e >> 5, k = e & 31;
    const float* p = gA + (size_t)r * lda + k;
    float4 x0 = *(const float4*)(p);
    float4 x1 = *(const float4*)(p + 4);
    __bf16* q = As + r * LDK + k;
    q[0] = (__bf16)x0.x; q[1] = (__bf16)x0.y; q[2] = (__bf16)x0.z; q[3] = (__bf16)x0.w;
    q[4] = (__bf16)x1.x; q[5] = (__bf16)x1.y; q[6] = (__bf16)x1.z; q[7] = (__bf16)x1.w;
}

// B tile: 128 n-rows x 32 k from bf16 global laid out [n][k] (k contiguous).
__device__ __forceinline__ void stage_b_rowk(__bf16* Bs, const __bf16* gB, int ldb, int tid) {
#pragma unroll
    for (int j = 0; j < 2; ++j) {
        int e = tid * 16 + j * 8;
        int r = e >> 5, k = e & 31;
        v8bf v = *(const v8bf*)(gB + (size_t)r * ldb + k);
        *(v8bf*)(Bs + r * LDK + k) = v;
    }
}

// B tile for h' GEMM: B[k=n][d]; source H is [n][d] -> transpose into [d][n].
__device__ __forceinline__ void stage_b_transpose(__bf16* Bs, const __bf16* gB, int ldb, int tid) {
#pragma unroll
    for (int j = 0; j < 2; ++j) {
        int c  = tid * 2 + j;        // 0..511
        int nl = c & 31;             // k (= n) index within tile
        int dl = (c >> 5) * 8;       // d chunk (0..120)
        v8bf v = *(const v8bf*)(gB + (size_t)nl * ldb + dl);
#pragma unroll
        for (int i = 0; i < 8; ++i) Bs[(dl + i) * LDK + nl] = v[i];
    }
}

// ---------------- conversion kernels ---------------------------------------
__global__ __launch_bounds__(256) void cvt_f32_bf16_kernel(const float* __restrict__ src,
                                                           __bf16* __restrict__ dst, int n) {
    for (int i = blockIdx.x * 256 + threadIdx.x; i < n; i += gridDim.x * 256)
        dst[i] = (__bf16)src[i];
}

// dst[c*rows + r] = bf16(src[r*cols + c])
__global__ __launch_bounds__(256) void transpose_cvt_kernel(const float* __restrict__ src,
                                                            __bf16* __restrict__ dst,
                                                            int rows, int cols) {
    int i = blockIdx.x * 256 + threadIdx.x;
    if (i < rows * cols) {
        int r = i / cols, c = i % cols;
        dst[(size_t)c * rows + r] = (__bf16)src[(size_t)r * cols + c];
    }
}

// ---------------- GEMM 1: RW = R * W_s  (bf16 out) -------------------------
__global__ __launch_bounds__(256)
void gemm_rw_kernel(const __bf16* __restrict__ Rb, const __bf16* __restrict__ Wt,
                    __bf16* __restrict__ RWb) {
    __shared__ __bf16 As[64 * LDK];
    __shared__ __bf16 Bs[128 * LDK];
    int tid = threadIdx.x, lane = tid & 31, w = tid >> 5;
    int wm = w & 1, wn = w >> 1;
    int m0 = blockIdx.x * 64, n0 = blockIdx.y * 128, b = blockIdx.z;
    const __bf16* Ag = Rb + ((size_t)b * Mc + m0) * Dc;
    const __bf16* Bg = Wt + (size_t)n0 * Dc;       // Wt: [d_out][d_in]
    v8f acc[4] = {};
    for (int kt = 0; kt < Dc; kt += 32) {
        __syncthreads();
        stage_a_bf16(As, Ag + kt, Dc, tid);
        stage_b_rowk(Bs, Bg + kt, Dc, tid);
        __syncthreads();
        mma_step(As, Bs, wm, wn, lane, acc);
    }
    int col = lane & 15, rb = (lane < 16) ? 0 : 8;
#pragma unroll
    for (int f = 0; f < 4; ++f) {
        int ms = wm * 32 + (f >> 1) * 16, ns = wn * 32 + (f & 1) * 16;
#pragma unroll
        for (int r = 0; r < 8; ++r) {
            size_t idx = ((size_t)b * Mc + m0 + ms + rb + r) * Dc + (n0 + ns + col);
            RWb[idx] = (__bf16)acc[f][r];
        }
    }
}

// ---------------- GEMM 2: S = RW * H^T, fused mask (fp32 out) --------------
__global__ __launch_bounds__(256)
void gemm_scores_kernel(const __bf16* __restrict__ RWb, const __bf16* __restrict__ Hb,
                        const int* __restrict__ mask, float* __restrict__ S) {
    __shared__ __bf16 As[64 * LDK];
    __shared__ __bf16 Bs[128 * LDK];
    int tid = threadIdx.x, lane = tid & 31, w = tid >> 5;
    int wm = w & 1, wn = w >> 1;
    int m0 = blockIdx.x * 64, n0 = blockIdx.y * 128, b = blockIdx.z;
    const __bf16* Ag = RWb + ((size_t)b * Mc + m0) * Dc;
    const __bf16* Bg = Hb  + ((size_t)b * Nc + n0) * Dc;   // H: [n][d], k-contiguous
    v8f acc[4] = {};
    for (int kt = 0; kt < Dc; kt += 32) {
        __syncthreads();
        stage_a_bf16(As, Ag + kt, Dc, tid);
        stage_b_rowk(Bs, Bg + kt, Dc, tid);
        __syncthreads();
        mma_step(As, Bs, wm, wn, lane, acc);
    }
    int col = lane & 15, rb = (lane < 16) ? 0 : 8;
#pragma unroll
    for (int f = 0; f < 4; ++f) {
        int ms = wm * 32 + (f >> 1) * 16, ns = wn * 32 + (f & 1) * 16;
        int n  = n0 + ns + col;
        int mk = mask[b * Nc + n];
#pragma unroll
        for (int r = 0; r < 8; ++r) {
            size_t idx = ((size_t)b * Mc + m0 + ms + rb + r) * Nc + n;
            S[idx] = mk ? acc[f][r] : NEGV;
        }
    }
}

// ---------------- softmax reductions ---------------------------------------
__device__ __forceinline__ float block_max(float v, float* sh, int tid) {
    sh[tid] = v; __syncthreads();
    for (int s = 128; s > 0; s >>= 1) { if (tid < s) sh[tid] = fmaxf(sh[tid], sh[tid + s]); __syncthreads(); }
    float r = sh[0]; __syncthreads(); return r;
}
__device__ __forceinline__ float block_sum(float v, float* sh, int tid) {
    sh[tid] = v; __syncthreads();
    for (int s = 128; s > 0; s >>= 1) { if (tid < s) sh[tid] += sh[tid + s]; __syncthreads(); }
    float r = sh[0]; __syncthreads(); return r;
}

__global__ __launch_bounds__(256)
void row_stats_kernel(const float* __restrict__ S, float* __restrict__ rmax, float* __restrict__ rsum) {
    __shared__ float sh[256];
    int bm = blockIdx.x, tid = threadIdx.x;
    const float* row = S + (size_t)bm * Nc;
    float mx = -3.0e38f, v[4];
#pragma unroll
    for (int j = 0; j < 4; ++j) { v[j] = row[tid + j * 256]; mx = fmaxf(mx, v[j]); }
    mx = block_max(mx, sh, tid);
    float sm = 0.f;
#pragma unroll
    for (int j = 0; j < 4; ++j) sm += __expf(v[j] - mx);
    sm = block_sum(sm, sh, tid);
    if (tid == 0) { rmax[bm] = mx; rsum[bm] = sm; }
}

__global__ __launch_bounds__(256)
void col_stats_kernel(const float* __restrict__ S, float* __restrict__ cmax, float* __restrict__ csum) {
    __shared__ float sh[256];
    int bn = blockIdx.x, tid = threadIdx.x;
    int b = bn >> 10, n = bn & (Nc - 1);
    const float* base = S + (size_t)b * Mc * Nc + n;
    float mx = -3.0e38f, v[4];
#pragma unroll
    for (int j = 0; j < 4; ++j) { v[j] = base[(size_t)(tid + j * 256) * Nc]; mx = fmaxf(mx, v[j]); }
    mx = block_max(mx, sh, tid);
    float sm = 0.f;
#pragma unroll
    for (int j = 0; j < 4; ++j) sm += __expf(v[j] - mx);
    sm = block_sum(sm, sh, tid);
    if (tid == 0) { cmax[bn] = mx; csum[bn] = sm; }
}

// a_bar = normalize_n( softmax_n(S) * softmax_m(S) ), stored bf16
__global__ __launch_bounds__(256)
void abar_kernel(const float* __restrict__ S,
                 const float* __restrict__ rmax, const float* __restrict__ rsum,
                 const float* __restrict__ cmax, const float* __restrict__ csum,
                 __bf16* __restrict__ Ab) {
    __shared__ float sh[256];
    int bm = blockIdx.x, tid = threadIdx.x;
    int b = bm >> 10;
    const float* row = S + (size_t)bm * Nc;
    float rm = rmax[bm], rsinv = 1.0f / rsum[bm];
    float p[4], local = 0.f;
#pragma unroll
    for (int j = 0; j < 4; ++j) {
        int n = tid + j * 256;
        float s  = row[n];
        float cm = cmax[b * Nc + n];
        float cs = csum[b * Nc + n];
        float a1 = __expf(s - rm) * rsinv;
        float a2 = __expf(s - cm) / cs;
        p[j] = a1 * a2;
        local += p[j];
    }
    float tot = block_sum(local, sh, tid);
    float inv = 1.0f / fmaxf(tot, 1e-8f);
#pragma unroll
    for (int j = 0; j < 4; ++j) Ab[(size_t)bm * Nc + tid + j * 256] = (__bf16)(p[j] * inv);
}

// ---------------- GEMM 3: h' = a_bar * H  (fp32 out) -----------------------
__global__ __launch_bounds__(256)
void gemm_hprime_kernel(const __bf16* __restrict__ Ab, const __bf16* __restrict__ Hb,
                        float* __restrict__ Hp) {
    __shared__ __bf16 As[64 * LDK];
    __shared__ __bf16 Bs[128 * LDK];
    int tid = threadIdx.x, lane = tid & 31, w = tid >> 5;
    int wm = w & 1, wn = w >> 1;
    int m0 = blockIdx.x * 64, d0 = blockIdx.y * 128, b = blockIdx.z;
    const __bf16* Ag = Ab + ((size_t)b * Mc + m0) * Nc;
    v8f acc[4] = {};
    for (int kt = 0; kt < Nc; kt += 32) {
        __syncthreads();
        stage_a_bf16(As, Ag + kt, Nc, tid);
        // B[k=n][d]: source H rows kt..kt+31, cols d0..d0+127, transposed into LDS
        stage_b_transpose(Bs, Hb + ((size_t)b * Nc + kt) * Dc + d0, Dc, tid);
        __syncthreads();
        mma_step(As, Bs, wm, wn, lane, acc);
    }
    int col = lane & 15, rb = (lane < 16) ? 0 : 8;
#pragma unroll
    for (int f = 0; f < 4; ++f) {
        int ms = wm * 32 + (f >> 1) * 16, ns = wn * 32 + (f & 1) * 16;
#pragma unroll
        for (int r = 0; r < 8; ++r) {
            size_t idx = ((size_t)b * Mc + m0 + ms + rb + r) * Dc + (d0 + ns + col);
            Hp[idx] = acc[f][r];
        }
    }
}

// ---------------- GEMM 4: gate + blend (fused epilogue) --------------------
// acc = R*Wg_top + h'*Wg_bot ; G = sigmoid(acc + b) ; out = G*R + (1-G)*h'
__global__ __launch_bounds__(256)
void gemm_gate_kernel(const __bf16* __restrict__ Rb, const float* __restrict__ Hp,
                      const __bf16* __restrict__ G1t, const __bf16* __restrict__ G2t,
                      const float* __restrict__ gate_b,
                      const float* __restrict__ R, float* __restrict__ out) {
    __shared__ __bf16 As[64 * LDK];
    __shared__ __bf16 Bs[128 * LDK];
    int tid = threadIdx.x, lane = tid & 31, w = tid >> 5;
    int wm = w & 1, wn = w >> 1;
    int m0 = blockIdx.x * 64, d0 = blockIdx.y * 128, b = blockIdx.z;
    const __bf16* Ag1 = Rb + ((size_t)b * Mc + m0) * Dc;
    const float*  Ag2 = Hp + ((size_t)b * Mc + m0) * Dc;
    v8f acc[4] = {};
    for (int kt = 0; kt < Dc; kt += 32) {          // pass 1: R * Wg_top
        __syncthreads();
        stage_a_bf16(As, Ag1 + kt, Dc, tid);
        stage_b_rowk(Bs, G1t + (size_t)d0 * Dc + kt, Dc, tid);
        __syncthreads();
        mma_step(As, Bs, wm, wn, lane, acc);
    }
    for (int kt = 0; kt < Dc; kt += 32) {          // pass 2: h' * Wg_bot (accumulate)
        __syncthreads();
        stage_a_f32(As, Ag2 + kt, Dc, tid);
        stage_b_rowk(Bs, G2t + (size_t)d0 * Dc + kt, Dc, tid);
        __syncthreads();
        mma_step(As, Bs, wm, wn, lane, acc);
    }
    int col = lane & 15, rb = (lane < 16) ? 0 : 8;
#pragma unroll
    for (int f = 0; f < 4; ++f) {
        int ms = wm * 32 + (f >> 1) * 16, ns = wn * 32 + (f & 1) * 16;
        int d  = d0 + ns + col;
        float gb = gate_b[d];
#pragma unroll
        for (int r = 0; r < 8; ++r) {
            size_t idx = ((size_t)b * Mc + m0 + ms + rb + r) * Dc + d;
            float pre = acc[f][r] + gb;
            float g   = 1.0f / (1.0f + __expf(-pre));
            float rv  = R[idx];
            float hv  = Hp[idx];
            out[idx]  = g * rv + (1.0f - g) * hv;
        }
    }
}

// ---------------------------------------------------------------------------
extern "C" void kernel_launch(void* const* d_in, const int* in_sizes, int n_in,
                              void* d_out, int out_size, void* d_ws, size_t ws_size,
                              hipStream_t stream) {
    (void)in_sizes; (void)n_in; (void)out_size; (void)ws_size;
    const float* R      = (const float*)d_in[0];
    const float* H      = (const float*)d_in[1];
    const float* W_s    = (const float*)d_in[2];
    const float* gate_w = (const float*)d_in[3];
    const float* gate_b = (const float*)d_in[4];
    const int*   mask   = (const int*)d_in[5];
    float* out = (float*)d_out;

    char* ws = (char*)d_ws;
    auto al = [](size_t x) { return (x + 255) & ~(size_t)255; };
    size_t off = 0;
    __bf16* Rb  = (__bf16*)(ws + off); off = al(off + (size_t)Bc * Mc * Dc * 2);
    __bf16* Hb  = (__bf16*)(ws + off); off = al(off + (size_t)Bc * Nc * Dc * 2);
    __bf16* Wt  = (__bf16*)(ws + off); off = al(off + (size_t)Dc * Dc * 2);
    __bf16* G1t = (__bf16*)(ws + off); off = al(off + (size_t)Dc * Dc * 2);
    __bf16* G2t = (__bf16*)(ws + off); off = al(off + (size_t)Dc * Dc * 2);
    __bf16* RWb = (__bf16*)(ws + off); off = al(off + (size_t)Bc * Mc * Dc * 2);
    float*  S   = (float*)(ws + off);  off = al(off + (size_t)Bc * Mc * Nc * 4);
    float*  rmx = (float*)(ws + off);  off = al(off + (size_t)Bc * Mc * 4);
    float*  rsm = (float*)(ws + off);  off = al(off + (size_t)Bc * Mc * 4);
    float*  cmx = (float*)(ws + off);  off = al(off + (size_t)Bc * Nc * 4);
    float*  csm = (float*)(ws + off);  off = al(off + (size_t)Bc * Nc * 4);
    __bf16* Ab  = (__bf16*)(ws + off); off = al(off + (size_t)Bc * Mc * Nc * 2);
    float*  Hp  = (float*)(ws + off);  off = al(off + (size_t)Bc * Mc * Dc * 4);

    const int nRD = Bc * Mc * Dc;   // 12582912
    const int nHD = Bc * Nc * Dc;
    const int nW  = Dc * Dc;

    cvt_f32_bf16_kernel<<<dim3((nRD + 255) / 256), 256, 0, stream>>>(R, Rb, nRD);
    cvt_f32_bf16_kernel<<<dim3((nHD + 255) / 256), 256, 0, stream>>>(H, Hb, nHD);
    transpose_cvt_kernel<<<dim3((nW + 255) / 256), 256, 0, stream>>>(W_s, Wt, Dc, Dc);
    transpose_cvt_kernel<<<dim3((nW + 255) / 256), 256, 0, stream>>>(gate_w, G1t, Dc, Dc);
    transpose_cvt_kernel<<<dim3((nW + 255) / 256), 256, 0, stream>>>(gate_w + (size_t)Dc * Dc, G2t, Dc, Dc);

    gemm_rw_kernel<<<dim3(Mc / 64, Dc / 128, Bc), 256, 0, stream>>>(Rb, Wt, RWb);
    gemm_scores_kernel<<<dim3(Mc / 64, Nc / 128, Bc), 256, 0, stream>>>(RWb, Hb, mask, S);

    row_stats_kernel<<<dim3(Bc * Mc), 256, 0, stream>>>(S, rmx, rsm);
    col_stats_kernel<<<dim3(Bc * Nc), 256, 0, stream>>>(S, cmx, csm);
    abar_kernel<<<dim3(Bc * Mc), 256, 0, stream>>>(S, rmx, rsm, cmx, csm, Ab);

    gemm_hprime_kernel<<<dim3(Mc / 64, Dc / 128, Bc), 256, 0, stream>>>(Ab, Hb, Hp);
    gemm_gate_kernel<<<dim3(Mc / 64, Dc / 128, Bc), 256, 0, stream>>>(Rb, Hp, G1t, G2t, gate_b, R, out);
}
